// BarrierNet_16432544874702
// MI455X (gfx1250) — matrix operand order repaired and checked by reading, not compiled, not run on hardware
//
#include <hip/hip_runtime.h>
#include <hip/hip_bf16.h>

typedef __attribute__((ext_vector_type(16))) _Float16 v16h;
typedef __attribute__((ext_vector_type(8)))  _Float16 v8h;
typedef __attribute__((ext_vector_type(8)))  float    v8f;

#define NF     128
#define NH1    128
#define NH2    32
#define NCLS   2
#define BATCH  16384

// f16 weight workspace offsets (in elements)
#define OFF_WIN  0
#define OFF_W1A  (128*128)
#define OFF_W2A  (OFF_W1A + 32*128)
#define OFF_W1B  (OFF_W2A + 128*32)
#define OFF_W2B  (OFF_W1B + 32*128)
#define OFF_WOUT (OFF_W2B + 128*32)          // padded to 16x128
#define OFF_UNOM_BYTES 131072                 // float u_nom[BATCH*2] here

// LDS row strides (halves); padded to avoid bank conflicts, keep 16B row align
#define LDX 136
#define LDH 40

__device__ inline v8f wmma16(v16h a, v16h b, v8f c) {
  return __builtin_amdgcn_wmma_f32_16x16x32_f16(
      /*neg_a=*/false, a, /*neg_b=*/false, b,
      /*c_mod=*/(short)0, c, /*reuse_a=*/false, /*reuse_b=*/false);
}

// 16-bit A-matrix 16x32 layout:
//  lanes 0-15 : M=lane,    elems 0..7 = K k0+0..7,  elems 8..15 = K k0+16..23
//  lanes 16-31: M=lane-16, elems 0..7 = K k0+8..15, elems 8..15 = K k0+24..31
__device__ inline v16h load_a16(const _Float16* base, int ld, int r0, int k0, int lane) {
  int m    = lane & 15;
  int koff = (lane & 16) ? 8 : 0;
  const _Float16* p = base + (size_t)(r0 + m) * ld + k0 + koff;
  v8h lo = *(const v8h*)(p);
  v8h hi = *(const v8h*)(p + 16);
  return __builtin_shufflevector(lo, hi, 0,1,2,3,4,5,6,7,8,9,10,11,12,13,14,15);
}

// Same A layout but converting from f32 source (obs)
__device__ inline v16h load_a32(const float* base, int ld, int r0, int k0, int lane) {
  int m    = lane & 15;
  int koff = (lane & 16) ? 8 : 0;
  const float* p = base + (size_t)(r0 + m) * ld + k0 + koff;
  v16h a = {};
#pragma unroll
  for (int t = 0; t < 8; ++t) a[t]     = (_Float16)p[t];
#pragma unroll
  for (int t = 0; t < 8; ++t) a[8 + t] = (_Float16)p[16 + t];
  return a;
}

// 16-bit B-matrix 32x16 layout: B[k][n] = W[n0+n][kbase+k] (row-major W, row len ld)
//  lanes 0-15 : N=lane,    elems 0..7 = K k0+0..7,  elems 8..15 = K k0+16..23
//  lanes 16-31: N=lane-16, elems 0..7 = K k0+8..15, elems 8..15 = K k0+24..31
__device__ inline v16h load_b16(const _Float16* W, int ld, int n0, int k0, int lane) {
  int n    = lane & 15;
  int koff = (lane & 16) ? 8 : 0;
  const _Float16* p = W + (size_t)(n0 + n) * ld + k0 + koff;
  v8h lo = *(const v8h*)(p);
  v8h hi = *(const v8h*)(p + 16);
  return __builtin_shufflevector(lo, hi, 0,1,2,3,4,5,6,7,8,9,10,11,12,13,14,15);
}

__global__ __launch_bounds__(256) void prep_weights(
    const float* W_in, const float* W1a, const float* W2a,
    const float* W1b, const float* W2b, const float* W_out, _Float16* wf) {
  int i = blockIdx.x * blockDim.x + threadIdx.x;
  if (i < 128 * 128) wf[OFF_WIN + i] = (_Float16)W_in[i];
  if (i < 32 * 128) { wf[OFF_W1A + i] = (_Float16)W1a[i]; wf[OFF_W1B + i] = (_Float16)W1b[i]; }
  if (i < 128 * 32) { wf[OFF_W2A + i] = (_Float16)W2a[i]; wf[OFF_W2B + i] = (_Float16)W2b[i]; }
  if (i < 16 * 128) wf[OFF_WOUT + i] = (i < NCLS * 128) ? (_Float16)W_out[i] : (_Float16)0.0f;
}

__global__ __launch_bounds__(128) void mlp_kernel(
    const float* __restrict__ obs, const _Float16* __restrict__ wf,
    const float* __restrict__ b_in,
    const float* __restrict__ b1a, const float* __restrict__ b2a,
    const float* __restrict__ b1b, const float* __restrict__ b2b,
    const float* __restrict__ b_out, float* __restrict__ u_nom) {
  __shared__ __align__(16) _Float16 xs[64 * LDX];
  __shared__ __align__(16) _Float16 hs[64 * LDH];

  const int wave   = threadIdx.x >> 5;          // 0..3, each wave owns 16 rows
  const int lane   = threadIdx.x & 31;
  const int lrow   = wave * 16;                 // LDS row base for this wave
  const int rowblk = blockIdx.x * 64 + lrow;    // global row base
  const int m      = lane & 15;
  const int madd   = (lane & 16) ? 8 : 0;       // D-matrix M offset for this lane half

  // ---- Layer 0: x = relu(obs @ W_in^T + b_in)   (K=128, N=128)
  v16h A0[4];
#pragma unroll
  for (int kk = 0; kk < 4; ++kk)
    A0[kk] = load_a32(obs, NF, rowblk, kk * 32, lane);

  for (int n0 = 0; n0 < NH1; n0 += 16) {
    v8f acc = {};
#pragma unroll
    for (int kk = 0; kk < 4; ++kk)
      acc = wmma16(A0[kk], load_b16(wf + OFF_WIN, NF, n0, kk * 32, lane), acc);
    float bias = b_in[n0 + m];
#pragma unroll
    for (int j = 0; j < 8; ++j) {
      int r = lrow + j + madd;
      xs[r * LDX + n0 + m] = (_Float16)fmaxf(acc[j] + bias, 0.0f);
    }
  }

  // ---- Two residual blocks: h = relu(x@W1^T+b1); x = relu(x + h@W2^T + b2)
  for (int blk = 0; blk < 2; ++blk) {
    const _Float16* W1 = wf + (blk == 0 ? OFF_W1A : OFF_W1B);
    const _Float16* W2 = wf + (blk == 0 ? OFF_W2A : OFF_W2B);
    const float* b1 = (blk == 0) ? b1a : b1b;
    const float* b2 = (blk == 0) ? b2a : b2b;

    for (int n0 = 0; n0 < NH2; n0 += 16) {
      v8f acc = {};
#pragma unroll
      for (int kk = 0; kk < 4; ++kk)
        acc = wmma16(load_a16(xs, LDX, lrow, kk * 32, lane),
                     load_b16(W1, NH1, n0, kk * 32, lane), acc);
      float bias = b1[n0 + m];
#pragma unroll
      for (int j = 0; j < 8; ++j) {
        int r = lrow + j + madd;
        hs[r * LDH + n0 + m] = (_Float16)fmaxf(acc[j] + bias, 0.0f);
      }
    }

    v16h Ah = load_a16(hs, LDH, lrow, 0, lane);   // K=32, one chunk
    for (int n0 = 0; n0 < NH1; n0 += 16) {
      v8f acc = {};
      acc = wmma16(Ah, load_b16(W2, NH2, n0, 0, lane), acc);
      float bias = b2[n0 + m];
#pragma unroll
      for (int j = 0; j < 8; ++j) {
        int r = lrow + j + madd;
        float xv = (float)xs[r * LDX + n0 + m];
        xs[r * LDX + n0 + m] = (_Float16)fmaxf(xv + acc[j] + bias, 0.0f);
      }
    }
  }

  // ---- Head: u_nom = x @ W_out^T + b_out  (N padded 2->16)
  {
    v8f acc = {};
#pragma unroll
    for (int kk = 0; kk < 4; ++kk)
      acc = wmma16(load_a16(xs, LDX, lrow, kk * 32, lane),
                   load_b16(wf + OFF_WOUT, NH1, 0, kk * 32, lane), acc);
    if (m < NCLS) {
      float bias = b_out[m];
#pragma unroll
      for (int j = 0; j < 8; ++j) {
        int r = rowblk + j + madd;
        u_nom[r * NCLS + m] = acc[j] + bias;
      }
    }
  }
}

__global__ __launch_bounds__(256) void qp_kernel(
    const float* __restrict__ obs, const float* __restrict__ u_nom,
    float* __restrict__ out) {
  int i = blockIdx.x * blockDim.x + threadIdx.x;
  if (i >= BATCH) return;
  const float* o = obs + (size_t)i * NF;
  double rx = (double)o[6], ry = (double)o[7];
  double vx = (double)o[8], vy = (double)o[9];
  double hcbf = rx * rx + ry * ry - 0.64;            // SAFE^2
  double lf = -2.0 * rx * vx - 2.0 * ry * vy;

  double G[6][3] = {
    {-2.0 * rx, -2.0 * ry, -1.0},
    { 0.0,  0.0, -1.0},
    { 1.0,  0.0,  0.0},
    { 0.0,  1.0,  0.0},
    {-1.0,  0.0,  0.0},
    { 0.0, -1.0,  0.0}};
  double hh[6] = { lf + 2.0 * hcbf, 0.0, 3.0, 3.0, 3.0, 3.0 };
  double Qd[3] = {2.0, 2.0, 20.0};
  double p[3]  = { -2.0 * (double)u_nom[2 * i],
                   -2.0 * (double)u_nom[2 * i + 1], 0.0 };
  double z[3] = {0.0, 0.0, 0.0};
  double s[6], lam[6];
#pragma unroll
  for (int k = 0; k < 6; ++k) { s[k] = 1.0; lam[k] = 1.0; }

  for (int it = 0; it < 40; ++it) {
    double smu = 0.0;
#pragma unroll
    for (int k = 0; k < 6; ++k) smu += lam[k] * s[k];
    smu *= (0.1 / 6.0);                               // SIGMA * mean(lam*s)

    double rp[6], ls[6], w[6];
#pragma unroll
    for (int k = 0; k < 6; ++k) {
      double gz = G[k][0] * z[0] + G[k][1] * z[1] + G[k][2] * z[2];
      rp[k] = gz + s[k] - hh[k];
      ls[k] = lam[k] / s[k];
      w[k]  = (smu - lam[k] * s[k] + lam[k] * rp[k]) / s[k];
    }
    double rd[3];
#pragma unroll
    for (int j = 0; j < 3; ++j) {
      double acc = Qd[j] * z[j] + p[j];
#pragma unroll
      for (int k = 0; k < 6; ++k) acc += G[k][j] * lam[k];
      rd[j] = acc;
    }
    double M[3][3];
#pragma unroll
    for (int a = 0; a < 3; ++a)
#pragma unroll
      for (int b = 0; b < 3; ++b) {
        double acc = (a == b) ? Qd[a] : 0.0;
#pragma unroll
        for (int k = 0; k < 6; ++k) acc += G[k][a] * ls[k] * G[k][b];
        M[a][b] = acc;
      }
    double rhs[3];
#pragma unroll
    for (int j = 0; j < 3; ++j) {
      double acc = rd[j];
#pragma unroll
      for (int k = 0; k < 6; ++k) acc += G[k][j] * w[k];
      rhs[j] = -acc;
    }
    // 3x3 solve via adjugate (M is SPD)
    double c00 = M[1][1]*M[2][2] - M[1][2]*M[2][1];
    double c01 = M[1][2]*M[2][0] - M[1][0]*M[2][2];
    double c02 = M[1][0]*M[2][1] - M[1][1]*M[2][0];
    double det = M[0][0]*c00 + M[0][1]*c01 + M[0][2]*c02;
    double inv = 1.0 / det;
    double c10 = M[0][2]*M[2][1] - M[0][1]*M[2][2];
    double c11 = M[0][0]*M[2][2] - M[0][2]*M[2][0];
    double c12 = M[0][1]*M[2][0] - M[0][0]*M[2][1];
    double c20 = M[0][1]*M[1][2] - M[0][2]*M[1][1];
    double c21 = M[0][2]*M[1][0] - M[0][0]*M[1][2];
    double c22 = M[0][0]*M[1][1] - M[0][1]*M[1][0];
    double dz0 = (c00*rhs[0] + c10*rhs[1] + c20*rhs[2]) * inv;
    double dz1 = (c01*rhs[0] + c11*rhs[1] + c21*rhs[2]) * inv;
    double dz2 = (c02*rhs[0] + c12*rhs[1] + c22*rhs[2]) * inv;

    double dlam[6], ds[6];
#pragma unroll
    for (int k = 0; k < 6; ++k) {
      double gdz = G[k][0]*dz0 + G[k][1]*dz1 + G[k][2]*dz2;
      dlam[k] = w[k] + ls[k] * gdz;
      ds[k]   = -rp[k] - gdz;
    }
    double step = 1e10;
#pragma unroll
    for (int k = 0; k < 6; ++k) {
      double r1 = (ds[k]   < 0.0) ? (-s[k]   / ds[k])   : 1e10;
      double r2 = (dlam[k] < 0.0) ? (-lam[k] / dlam[k]) : 1e10;
      step = fmin(step, fmin(r1, r2));
    }
    double a = fmin(1.0, 0.99 * step);
    z[0] += a * dz0; z[1] += a * dz1; z[2] += a * dz2;
#pragma unroll
    for (int k = 0; k < 6; ++k) { s[k] += a * ds[k]; lam[k] += a * dlam[k]; }
  }
  out[2 * i + 0] = (float)z[0];
  out[2 * i + 1] = (float)z[1];
}

extern "C" void kernel_launch(void* const* d_in, const int* in_sizes, int n_in,
                              void* d_out, int out_size, void* d_ws, size_t ws_size,
                              hipStream_t stream) {
  const float* obs   = (const float*)d_in[0];
  const float* W_in  = (const float*)d_in[1];
  const float* b_in  = (const float*)d_in[2];
  const float* W1a   = (const float*)d_in[3];
  const float* b1a   = (const float*)d_in[4];
  const float* W2a   = (const float*)d_in[5];
  const float* b2a   = (const float*)d_in[6];
  const float* W1b   = (const float*)d_in[7];
  const float* b1b   = (const float*)d_in[8];
  const float* W2b   = (const float*)d_in[9];
  const float* b2b   = (const float*)d_in[10];
  const float* W_out = (const float*)d_in[11];
  const float* b_out = (const float*)d_in[12];

  _Float16* wf    = (_Float16*)d_ws;
  float*    u_nom = (float*)((char*)d_ws + OFF_UNOM_BYTES);
  float*    out   = (float*)d_out;

  prep_weights<<<64, 256, 0, stream>>>(W_in, W1a, W2a, W1b, W2b, W_out, wf);
  mlp_kernel<<<BATCH / 64, 128, 0, stream>>>(obs, wf, b_in, b1a, b2a, b1b, b2b, b_out, u_nom);
  qp_kernel<<<BATCH / 256, 256, 0, stream>>>(obs, u_nom, out);
}
